// MDTA_24120536335141
// MI455X (gfx1250) — matrix operand order repaired and checked by reading, not compiled, not run on hardware
//
#include <hip/hip_runtime.h>

// ---------------------------------------------------------------------------
// MDTA (Restormer channel attention) for MI455X / gfx1250, wave32 + WMMA bf16
// ---------------------------------------------------------------------------

#define HW   16384   // 128*128
#define HDIM 128

typedef __attribute__((ext_vector_type(16))) __bf16 v16bf;
typedef __attribute__((ext_vector_type(8)))  float  v8f;

union BF16Frag {
    v16bf        v;
    unsigned int u[8];
};

__device__ __forceinline__ unsigned short f2bf(float f) {
    unsigned int u = __float_as_uint(f);
    unsigned int r = u + 0x7FFFu + ((u >> 16) & 1u);   // round-to-nearest-even
    return (unsigned short)(r >> 16);
}
__device__ __forceinline__ unsigned int pack2bf(float a, float b) {
    return (unsigned int)f2bf(a) | ((unsigned int)f2bf(b) << 16);
}

// ---------------------------------------------------------------------------
// Generic 1x1 conv as per-batch GEMM: out[b][m][n] = sum_c W[m][c]*in[b][c][n]
// grid: (N/128, M/64, B), block 256 (8 waves). Wave tile: 16M x 64N
// (4 accumulators -> 4 WMMAs per K-step sharing one A fragment).
// ---------------------------------------------------------------------------
__global__ __launch_bounds__(256) void conv1x1_wmma(
    const float* __restrict__ in,   // [B][K][HW]
    const float* __restrict__ W,    // [M][K]
    float*       __restrict__ out,  // [B][M][HW]
    int M, int K)
{
    const int b     = blockIdx.z;
    const int n_blk = blockIdx.x * 128;
    const int m_blk = blockIdx.y * 64;
    const int tid   = threadIdx.x;
    const int wave  = tid >> 5;
    const int lane  = tid & 31;
    const int l15   = lane & 15;
    const int lh    = lane >> 4;

    __shared__ __align__(16) unsigned short lds_a[64 * 34];   // [m][k] bf16 (pad 34)
    __shared__ __align__(16) unsigned short lds_b[128 * 34];  // [n][k] bf16 (transposed)

    const int mrow   = wave & 3;   // 4 M sub-tiles
    const int ngroup = wave >> 2;  // 2 groups of 64 N

    v8f acc[4] = {{}, {}, {}, {}};

    const float* inb = in + (size_t)b * K * HW;

    // staging addresses (constant across K loop except kk)
    const int a_m  = tid >> 2;         // 0..63
    const int a_k0 = (tid & 3) * 8;    // 0,8,16,24
    const int b_k  = tid >> 3;         // 0..31
    const int b_n0 = (tid & 7) * 16;   // 0..112

    for (int kk = 0; kk < K; kk += 32) {
        __syncthreads();
        // ---- Stage A: W[m_blk+m][kk+k] -> lds_a[m][k], wide b128 loads ----
        {
            const float* src = W + (size_t)(m_blk + a_m) * K + kk + a_k0;
            float4 p0 = *(const float4*)(src + 0);
            float4 p1 = *(const float4*)(src + 4);
            unsigned int* dst = (unsigned int*)&lds_a[a_m * 34 + a_k0];
            dst[0] = pack2bf(p0.x, p0.y);
            dst[1] = pack2bf(p0.z, p0.w);
            dst[2] = pack2bf(p1.x, p1.y);
            dst[3] = pack2bf(p1.z, p1.w);
            if (kk + 32 < K) __builtin_prefetch(src + 32, 0, 1);
        }
        // ---- Stage B transposed: in[b][kk+k][n_blk+n] -> lds_b[n][k] ----
        {
            const float* src = inb + (size_t)(kk + b_k) * HW + n_blk + b_n0;
            #pragma unroll
            for (int q = 0; q < 4; ++q) {
                float4 p = *(const float4*)(src + 4 * q);
                lds_b[(b_n0 + 4 * q + 0) * 34 + b_k] = f2bf(p.x);
                lds_b[(b_n0 + 4 * q + 1) * 34 + b_k] = f2bf(p.y);
                lds_b[(b_n0 + 4 * q + 2) * 34 + b_k] = f2bf(p.z);
                lds_b[(b_n0 + 4 * q + 3) * 34 + b_k] = f2bf(p.w);
            }
            if (kk + 32 < K) __builtin_prefetch(src + 32 * HW, 0, 1);
        }
        __syncthreads();

        // ---- A fragment (shared by 4 WMMAs) ----
        BF16Frag fa;
        {
            const unsigned short* ap = &lds_a[(mrow * 16 + l15) * 34];
            const int ka = lh * 8;
            #pragma unroll
            for (int j = 0; j < 4; ++j) {
                fa.u[j]     = *(const unsigned int*)&ap[ka + 2 * j];
                fa.u[4 + j] = *(const unsigned int*)&ap[16 + ka + 2 * j];
            }
        }
        // ---- 4 N sub-tiles ----
        const int kb = lh * 8;
        #pragma unroll
        for (int ns = 0; ns < 4; ++ns) {
            BF16Frag fb;
            const unsigned short* bp = &lds_b[(ngroup * 64 + ns * 16 + l15) * 34];
            #pragma unroll
            for (int j = 0; j < 8; ++j)
                fb.u[j] = *(const unsigned int*)&bp[kb + 2 * j + ((j & 4) << 1)];
            acc[ns] = __builtin_amdgcn_wmma_f32_16x16x32_bf16(false, fa.v, false, fb.v,
                                                              (short)0, acc[ns], false, false);
        }
    }

    // C/D layout: lane -> N = l15, VGPR r -> M = r + 8*lh
    float* outb = out + (size_t)b * M * HW;
    const int mbase = m_blk + mrow * 16 + lh * 8;
    #pragma unroll
    for (int ns = 0; ns < 4; ++ns) {
        const int n = n_blk + ngroup * 64 + ns * 16 + l15;
        #pragma unroll
        for (int r = 0; r < 8; ++r)
            outb[(size_t)(mbase + r) * HW + n] = acc[ns][r];
    }
}

// ---------------------------------------------------------------------------
// Depthwise 3x3, padding SAME. One thread per output element.
// ---------------------------------------------------------------------------
__global__ __launch_bounds__(256) void dwconv3x3_k(
    const float* __restrict__ in,  // [B][C][128][128]
    const float* __restrict__ w,   // [C][1][3][3]
    float*       __restrict__ out,
    int C)
{
    const int idx  = blockIdx.x * 256 + threadIdx.x;     // < B*C*HW (< 2^31)
    const int n    = idx & (HW - 1);
    const int rest = idx >> 14;                          // b*C + ch
    const int ch   = rest % C;
    const int hx   = n & (HDIM - 1);
    const int hy   = n >> 7;

    const float* ip = in + ((size_t)rest << 14);
    const float* wp = w + ch * 9;

    float s = 0.0f;
    #pragma unroll
    for (int dy = -1; dy <= 1; ++dy) {
        const int yy = hy + dy;
        if (yy < 0 || yy >= HDIM) continue;
        #pragma unroll
        for (int dx = -1; dx <= 1; ++dx) {
            const int xx = hx + dx;
            if (xx < 0 || xx >= HDIM) continue;
            s += ip[yy * HDIM + xx] * wp[(dy + 1) * 3 + (dx + 1)];
        }
    }
    out[idx] = s;
}

// ---------------------------------------------------------------------------
// Attention scores per (b,head): 48x48 gram over K=16384 via WMMA bf16,
// L2-norm scaling folded in post-hoc, softmax in LDS.
// grid 32 (= B*NH), block 288 (9 waves = 3x3 output tiles).
// ---------------------------------------------------------------------------
__global__ __launch_bounds__(288) void attn_scores(
    const float* __restrict__ q,    // [B][192][HW]  (channel = head*48 + c)
    const float* __restrict__ kv,   // [B][384][HW]  (k = ch 0..191, v = 192..383)
    const float* __restrict__ temp, // [4]
    float*       __restrict__ attn) // [32][48][48]
{
    const int bh   = blockIdx.x;          // b*4 + head
    const int head = bh & 3;
    const int bb   = bh >> 2;
    const int tid  = threadIdx.x;
    const int wv   = tid >> 5;
    const int lane = tid & 31;
    const int l15  = lane & 15;
    const int lh   = lane >> 4;

    __shared__ float inv_n[96];
    __shared__ float sc[48 * 49];

    const float* qb = q  + (size_t)bh * 48 * HW;
    const float* kb = kv + ((size_t)bb * 384 + head * 48) * HW;

    // --- L2 norms (one wave per row, strided rows) ---
    for (int r = wv; r < 96; r += 9) {
        const float* p = (r < 48) ? (qb + (size_t)r * HW)
                                  : (kb + (size_t)(r - 48) * HW);
        float s = 0.0f;
        for (int n = lane; n < HW; n += 32) { float v = p[n]; s += v * v; }
        #pragma unroll
        for (int off = 16; off; off >>= 1) s += __shfl_xor(s, off, 32);
        if (lane == 0) inv_n[r] = 1.0f / fmaxf(sqrtf(s), 1e-12f);
    }
    __syncthreads();

    // --- gram = q . k^T  (K dim = HW) ---
    const int mt = wv / 3, nt = wv % 3;
    v8f acc = {};
    const float* qrow = qb + (size_t)(mt * 16 + l15) * HW;
    const float* krow = kb + (size_t)(nt * 16 + l15) * HW;
    const int ka = lh * 8;
    const int kB = lh * 8;

    for (int kk = 0; kk < HW; kk += 32) {
        BF16Frag fa, fb;
        #pragma unroll
        for (int j = 0; j < 4; ++j) {
            float2 p0 = *(const float2*)&qrow[kk + ka + 2 * j];
            float2 p1 = *(const float2*)&qrow[kk + 16 + ka + 2 * j];
            fa.u[j]     = pack2bf(p0.x, p0.y);
            fa.u[4 + j] = pack2bf(p1.x, p1.y);
        }
        #pragma unroll
        for (int j = 0; j < 8; ++j) {
            float2 p = *(const float2*)&krow[kk + kB + 2 * j + ((j & 4) << 1)];
            fb.u[j] = pack2bf(p.x, p.y);
        }
        acc = __builtin_amdgcn_wmma_f32_16x16x32_bf16(false, fa.v, false, fb.v,
                                                      (short)0, acc, false, false);
    }

    // scale by inv-norms and temperature, stash to LDS
    const float t  = temp[head];
    const int   M0 = mt * 16 + lh * 8;
    const int   N  = nt * 16 + l15;
    #pragma unroll
    for (int r = 0; r < 8; ++r) {
        const int M = M0 + r;
        sc[M * 49 + N] = acc[r] * inv_n[M] * inv_n[48 + N] * t;
    }
    __syncthreads();

    // --- softmax over last dim, 48 rows by threads 0..47 ---
    if (tid < 48) {
        float* row = &sc[tid * 49];
        float mx = -1e30f;
        for (int j = 0; j < 48; ++j) mx = fmaxf(mx, row[j]);
        float s = 0.0f;
        for (int j = 0; j < 48; ++j) { float e = __expf(row[j] - mx); row[j] = e; s += e; }
        const float inv = 1.0f / s;
        float* o = attn + (size_t)bh * 48 * 48 + tid * 48;
        for (int j = 0; j < 48; ++j) o[j] = row[j] * inv;
    }
}

// ---------------------------------------------------------------------------
// out[48][HW] = attn[48][48] . v[48][HW], K zero-padded to 64 for WMMA.
// grid (HW/128, 32), block 256 (8 waves, one 16-wide N column each, 3 M tiles).
// ---------------------------------------------------------------------------
__global__ __launch_bounds__(256) void attn_apply(
    const float* __restrict__ attn, // [32][48][48]
    const float* __restrict__ kv,   // [B][384][HW]
    float*       __restrict__ out)  // [B][192][HW]
{
    const int bh   = blockIdx.y;
    const int head = bh & 3;
    const int bb   = bh >> 2;
    const int tid  = threadIdx.x;
    const int wv   = tid >> 5;
    const int lane = tid & 31;
    const int l15  = lane & 15;
    const int lh   = lane >> 4;

    __shared__ __align__(16) unsigned short la[48 * 66];  // attn bf16, K padded to 64

    const float* ap = attn + (size_t)bh * 48 * 48;
    for (int idx = tid; idx < 48 * 64; idx += 256) {
        const int m = idx >> 6, kq = idx & 63;
        const float val = (kq < 48) ? ap[m * 48 + kq] : 0.0f;
        la[m * 66 + kq] = f2bf(val);
    }
    __syncthreads();

    const float* vb = kv + ((size_t)bb * 384 + 192 + head * 48) * HW;
    const int n0 = blockIdx.x * 128 + wv * 16 + l15;

    v8f acc[3] = {{}, {}, {}};
    #pragma unroll
    for (int kk = 0; kk < 64; kk += 32) {
        BF16Frag fb;
        const int kB = kk + lh * 8;
        #pragma unroll
        for (int j = 0; j < 8; ++j) {
            const int K0 = kB + 2 * j + ((j & 4) << 1);
            const float a0 = (K0 < 48)     ? vb[(size_t)K0 * HW + n0]       : 0.0f;
            const float a1 = (K0 + 1 < 48) ? vb[(size_t)(K0 + 1) * HW + n0] : 0.0f;
            fb.u[j] = pack2bf(a0, a1);
        }
        #pragma unroll
        for (int mt = 0; mt < 3; ++mt) {
            BF16Frag fa;
            const unsigned short* arow = &la[(mt * 16 + l15) * 66 + kk];
            const int ka = lh * 8;
            #pragma unroll
            for (int j = 0; j < 4; ++j) {
                fa.u[j]     = *(const unsigned int*)&arow[ka + 2 * j];
                fa.u[4 + j] = *(const unsigned int*)&arow[16 + ka + 2 * j];
            }
            acc[mt] = __builtin_amdgcn_wmma_f32_16x16x32_bf16(false, fa.v, false, fb.v,
                                                              (short)0, acc[mt], false, false);
        }
    }

    float* ob = out + (size_t)bh * 48 * HW;
    #pragma unroll
    for (int mt = 0; mt < 3; ++mt) {
        const int M0 = mt * 16 + lh * 8;
        #pragma unroll
        for (int r = 0; r < 8; ++r)
            ob[(size_t)(M0 + r) * HW + n0] = acc[mt][r];
    }
}

// ---------------------------------------------------------------------------
// Host-side orchestration (graph-capture safe: launches only).
// ---------------------------------------------------------------------------
extern "C" void kernel_launch(void* const* d_in, const int* in_sizes, int n_in,
                              void* d_out, int out_size, void* d_ws, size_t ws_size,
                              hipStream_t stream) {
    (void)in_sizes; (void)n_in; (void)out_size; (void)ws_size;

    const float* x           = (const float*)d_in[0];  // [8,192,128,128]
    const float* y           = (const float*)d_in[1];  // [8,192,128,128]
    const float* w_qkv       = (const float*)d_in[2];  // [384,192]
    const float* w_qkv_dw    = (const float*)d_in[3];  // [384,1,3,3]
    const float* w_query     = (const float*)d_in[4];  // [192,192]
    const float* w_query_dw  = (const float*)d_in[5];  // [192,1,3,3]
    const float* w_proj      = (const float*)d_in[6];  // [192,192]
    const float* temperature = (const float*)d_in[7];  // [1,4,1,1]
    float* out = (float*)d_out;

    const int B = 8;
    const size_t n384 = (size_t)B * 384 * HW;  // 50.3M floats
    const size_t n192 = (size_t)B * 192 * HW;  // 25.2M floats

    float* buf0 = (float*)d_ws;        // kv_pre, later reused for q
    float* buf1 = buf0 + n384;         // kv (k | v)
    float* buf2 = buf1 + n384;         // q_pre, later reused for attn_out
    float* buf4 = buf2 + n192;         // attn probs [32,48,48]

    // 1) kv_pre = conv1x1(x, w_qkv)           [8,384,HW]
    conv1x1_wmma<<<dim3(HW / 128, 384 / 64, B), 256, 0, stream>>>(x, w_qkv, buf0, 384, 192);
    // 2) kv = dwconv3x3(kv_pre)
    dwconv3x3_k<<<dim3((unsigned)(n384 / 256)), 256, 0, stream>>>(buf0, w_qkv_dw, buf1, 384);
    // 3) q_pre = conv1x1(y, w_query)          [8,192,HW]
    conv1x1_wmma<<<dim3(HW / 128, 192 / 64, B), 256, 0, stream>>>(y, w_query, buf2, 192, 192);
    // 4) q = dwconv3x3(q_pre)  (into buf0, kv_pre is dead)
    dwconv3x3_k<<<dim3((unsigned)(n192 / 256)), 256, 0, stream>>>(buf2, w_query_dw, buf0, 192);
    // 5) attn = softmax(l2norm(q) . l2norm(k)^T * temperature)
    attn_scores<<<dim3(32), 288, 0, stream>>>(buf0, buf1, temperature, buf4);
    // 6) attn_out = attn . v   (into buf2, q_pre is dead)
    attn_apply<<<dim3(HW / 128, 32), 256, 0, stream>>>(buf4, buf1, buf2);
    // 7) out = conv1x1(attn_out, w_proj)
    conv1x1_wmma<<<dim3(HW / 128, 192 / 64, B), 256, 0, stream>>>(buf2, w_proj, out, 192, 192);
}
